// TissueABMIL_22179211116840
// MI455X (gfx1250) — matrix operand order repaired and checked by reading, not compiled, not run on hardware
//
#include <hip/hip_runtime.h>
#include <hip/hip_bf16.h>
#include <math.h>
#include <stdint.h>

#define B_   8
#define N_   8192
#define D_   1536
#define H_   4
#define HD_  256
#define T_   50
#define E_   64
#define NCOL 2048            // H*HD (att) + H*HD (gate)
#define LDS_ROW (D_ + 8)     // A-tile row stride: +8 bf16 pad -> 4-bank shift/row
#define TILE_M 64
#define KCHUNK 256           // B staged in 6 K-chunks of 256 (double-buffered)
#define NKCH  (D_ / KCHUNK)  // 6
#define BROW (KCHUNK + 8)    // B-slice column stride (264 bf16 = 132 dw -> 4-bank shift)
#define NCH  8               // N-chunks for pooling partials

typedef __bf16 bf16;
typedef __attribute__((ext_vector_type(16))) __bf16 v16bf;
typedef __attribute__((ext_vector_type(8)))  __bf16 v8bf;
typedef __attribute__((ext_vector_type(4)))  __bf16 v4bf;
typedef __attribute__((ext_vector_type(8)))  float  v8f;
typedef __attribute__((ext_vector_type(4)))  float  v4f;

// ---------------------------------------------------------------------------
// CDNA5 async global->LDS copy (ASYNCcnt-tracked DMA, no VGPR round trip).
// Generic LDS pointer: addr[31:0] is the LDS byte offset (ISA 10.2 aperture
// rules), which is exactly what the instruction's VDST operand wants.
// ---------------------------------------------------------------------------
__device__ __forceinline__ void async_copy16(void* lds_dst, const void* gsrc) {
    asm volatile("global_load_async_to_lds_b128 %0, %1, off"
                 :: "v"((unsigned)(uintptr_t)lds_dst),
                    "v"((unsigned long long)(uintptr_t)gsrc)
                 : "memory");
}
#define S_WAIT_ASYNC(n) asm volatile("s_wait_asynccnt %0" :: "n"(n) : "memory")

// ---------------------------------------------------------------------------
// Fast activations: CDNA5 has a hardware V_TANH_F32 transcendental.
// ---------------------------------------------------------------------------
__device__ __forceinline__ float fast_tanh(float x) {
#if __has_builtin(__builtin_amdgcn_tanhf)
    return __builtin_amdgcn_tanhf(x);
#else
    float xc = fminf(fmaxf(x, -10.0f), 10.0f);
    float e  = __expf(2.0f * xc);
    return (e - 1.0f) * __builtin_amdgcn_rcpf(e + 1.0f);
#endif
}
__device__ __forceinline__ float fast_sigmoid(float x) {
    return __builtin_amdgcn_rcpf(1.0f + __expf(-x));
}
__device__ __forceinline__ float gelu_f(float x) {
    return 0.5f * x * (1.0f + erff(x * 0.70710678118654752440f));
}

// ---------------------------------------------------------------------------
// Fragment loaders per CDNA5 wave32 WMMA VGPR layouts (CDNA5 ISA §7.12.2)
// A 16x32 bf16: lane L -> row (L&15); K chunks {k0+(L>>4)*8 .. +8} and +16
// B 32x16 bf16: lane L -> col (L&15); K chunks {k0+(L>>4)*8 .. +8} and +16
// ---------------------------------------------------------------------------
__device__ __forceinline__ v16bf load_a_frag(const bf16* base, int lane, int k0) {
    const int row = lane & 15;
    const int kb  = k0 + ((lane >> 4) << 3);
    const v8bf lo = *(const v8bf*)(base + row * LDS_ROW + kb);
    const v8bf hi = *(const v8bf*)(base + row * LDS_ROW + kb + 16);
    v16bf r;
#pragma unroll
    for (int i = 0; i < 8; ++i) { r[i] = lo[i]; r[i + 8] = hi[i]; }
    return r;
}

__device__ __forceinline__ v16bf load_b_frag(const bf16* base, int lane, int k0) {
    const int col = lane & 15;
    const int kb  = k0 + ((lane >> 4) << 3);
    const v8bf lo = *(const v8bf*)(base + col * BROW + kb);
    const v8bf hi = *(const v8bf*)(base + col * BROW + kb + 16);
    v16bf r;
#pragma unroll
    for (int i = 0; i < 8; ++i) { r[i] = lo[i]; r[i + 8] = hi[i]; }
    return r;
}

__device__ __forceinline__ v8f wmma_bf16(v16bf a, v16bf b, v8f c) {
    return __builtin_amdgcn_wmma_f32_16x16x32_bf16(false, a, false, b, (short)0, c,
                                                   false, false);
}

// ---------------------------------------------------------------------------
// K0: pack attW||gateW (H,D,HD) f32 -> bf16 column-major Wb[col][d]
// ---------------------------------------------------------------------------
__global__ void __launch_bounds__(256)
k_prep_weights(const float* __restrict__ attW, const float* __restrict__ gateW,
               bf16* __restrict__ Wb) {
    int idx = blockIdx.x * 256 + threadIdx.x;      // over NCOL*D_
    if (idx >= NCOL * D_) return;
    int j = idx / D_;
    int d = idx % D_;
    float v;
    if (j < H_ * HD_) {
        int h = j >> 8, k = j & 255;
        v = attW[((size_t)h * D_ + d) * HD_ + k];
    } else {
        int jj = j - H_ * HD_;
        int h = jj >> 8, k = jj & 255;
        v = gateW[((size_t)h * D_ + d) * HD_ + k];
    }
    Wb[(size_t)j * D_ + d] = (bf16)v;
}

// ---------------------------------------------------------------------------
// K1: fused gated-attention scores (the 4.1e11-FLOP GEMM).
//   pre_att = F @ Watt, pre_gate = F @ Wgate  (bf16 WMMA, f32 acc)
//   score[b,h,n] = sum_k tanh(pre_att+attb)*sigmoid(pre_gate+gateb)*brW + brb
// A tile (64x1536 bf16) staged once in LDS. Per step: 2 att + 2 gate 16-col
// tiles (one A frag feeds 4 WMMAs). B slices DMA'd via async-to-LDS in
// double-buffered 256-wide K-chunks; the K loop is software-pipelined.
// LDS is declared STATICALLY (~259 KB of the 320 KB WGP pool) so the backend
// can see that occupancy is 1 WG/WGP = 1 wave/SIMD: the scheduler then has no
// pressure-driven reason to collapse the fragment prefetch pipeline, and all
// DS addressing folds into immediate offsets.
// ---------------------------------------------------------------------------
__global__ void __launch_bounds__(128, 1) __attribute__((amdgpu_waves_per_eu(1, 1)))
k_scores(const float* __restrict__ features,
         const unsigned char* __restrict__ mask,
         const bf16* __restrict__ Wb,
         const float* __restrict__ attb,  const float* __restrict__ gateb,
         const float* __restrict__ brW,   const float* __restrict__ brb,
         float* __restrict__ ws_scores,   float* __restrict__ out_scores) {
    __shared__ bf16 sA[TILE_M][LDS_ROW];      // 193 KB
    __shared__ bf16 sB[2][64][BROW];          // 2 x 33 KB

    const int tid  = threadIdx.x;
    const int lane = tid & 31;
    const int wid  = tid >> 5;
    const int b    = blockIdx.x / (N_ / TILE_M);
    const int tile = blockIdx.x % (N_ / TILE_M);
    const int row0 = tile * TILE_M;

    // Stage A tile: convert f32 -> bf16 into LDS (features read exactly once).
    // 2 threads per row; pure power-of-two addressing, pointers strided.
    {
        const int row = tid >> 1, sub = tid & 1;
        const float* src = features + ((size_t)b * N_ + row0 + row) * D_ + sub * 4;
        bf16* dst = &sA[row][sub * 4];
#pragma unroll 4
        for (int it = 0; it < D_ / 8; ++it) {
            v4f v = *(const v4f*)src;
            v4bf o; o[0] = (bf16)v[0]; o[1] = (bf16)v[1]; o[2] = (bf16)v[2]; o[3] = (bf16)v[3];
            *(v4bf*)dst = o;
            src += 8; dst += 8;
        }
    }

    const int waveRow = wid * 16;
    const bf16* aBase = &sA[waveRow][0];

    // Issue one K-chunk stage: 64 cols x 256 k bf16 = 2048 x 16B transfers;
    // 1 thread owns 1 column pairwise -> exactly 16 wave-wide async ops/wave.
    // Buffer cols 0..31 = att cols ct2*32.., cols 32..63 = gate cols.
    auto stage_chunk = [&](int buf, int h, int ct2, int c) {
        const int col = tid >> 1;          // 0..63
        const int sub = tid & 1;
        const int gcol = (col < 32) ? (h * HD_ + ct2 * 32 + col)
                                    : (H_ * HD_ + h * HD_ + ct2 * 32 + (col - 32));
        const bf16* src = Wb + (size_t)gcol * D_ + c * KCHUNK + sub * 8;
        bf16* d = &sB[buf][col][sub * 8];
#pragma unroll
        for (int it = 0; it < KCHUNK / 16; ++it) {   // 16 iters, stride 32B
            async_copy16(d, src);
            d += 16; src += 16;
        }
    };

#pragma unroll 1
    for (int h = 0; h < H_; ++h) {
        float sacc[8] = {0, 0, 0, 0, 0, 0, 0, 0};
#pragma unroll 1
        for (int ct2 = 0; ct2 < 8; ++ct2) {          // 2 column-tiles per step
            stage_chunk(0, h, ct2, 0);

            v8f accA0 = {0, 0, 0, 0, 0, 0, 0, 0};
            v8f accA1 = {0, 0, 0, 0, 0, 0, 0, 0};
            v8f accG0 = {0, 0, 0, 0, 0, 0, 0, 0};
            v8f accG1 = {0, 0, 0, 0, 0, 0, 0, 0};
#pragma unroll 2
            for (int c = 0; c < NKCH; ++c) {
                if (c < NKCH - 1) {
                    stage_chunk((c + 1) & 1, h, ct2, c + 1);
                    S_WAIT_ASYNC(16);   // in-order retire: chunk c has landed
                } else {
                    S_WAIT_ASYNC(0);
                }
                __syncthreads();        // all waves' chunk-c DMA visible

                const bf16* bb = &sB[c & 1][0][0];
                // Software-pipelined K loop: prefetch ks+1 frags before WMMAs.
                v16bf a  = load_a_frag(aBase, lane, c * KCHUNK);
                v16bf b0 = load_b_frag(bb +  0 * BROW, lane, 0);
                v16bf b1 = load_b_frag(bb + 16 * BROW, lane, 0);
                v16bf b2 = load_b_frag(bb + 32 * BROW, lane, 0);
                v16bf b3 = load_b_frag(bb + 48 * BROW, lane, 0);
#pragma unroll
                for (int ks = 0; ks < KCHUNK / 32; ++ks) {
                    v16bf an, b0n, b1n, b2n, b3n;
                    if (ks < KCHUNK / 32 - 1) {
                        const int kl = (ks + 1) * 32;
                        an  = load_a_frag(aBase, lane, c * KCHUNK + kl);
                        b0n = load_b_frag(bb +  0 * BROW, lane, kl);
                        b1n = load_b_frag(bb + 16 * BROW, lane, kl);
                        b2n = load_b_frag(bb + 32 * BROW, lane, kl);
                        b3n = load_b_frag(bb + 48 * BROW, lane, kl);
                    }
                    accA0 = wmma_bf16(a, b0, accA0);
                    accA1 = wmma_bf16(a, b1, accA1);
                    accG0 = wmma_bf16(a, b2, accG0);
                    accG1 = wmma_bf16(a, b3, accG1);
                    if (ks < KCHUNK / 32 - 1) { a = an; b0 = b0n; b1 = b1n; b2 = b2n; b3 = b3n; }
                }
                __syncthreads();        // done reading buf before it is restaged
            }

            // Epilogue for both tiles: column -> k index; fuse tanh*sigmoid*brW.
            const int colL = h * HD_ + ct2 * 32 + (lane & 15);
            const int colH = colL + 16;
            const float baL = attb[colL], bgL = gateb[colL], brL = brW[colL];
            const float baH = attb[colH], bgH = gateb[colH], brH = brW[colH];
#pragma unroll
            for (int r = 0; r < 8; ++r) {
                sacc[r] += fast_tanh(accA0[r] + baL) * fast_sigmoid(accG0[r] + bgL) * brL
                         + fast_tanh(accA1[r] + baH) * fast_sigmoid(accG1[r] + bgH) * brH;
            }
        }
        // Reduce over the 16 columns held by each 16-lane half.
#pragma unroll
        for (int r = 0; r < 8; ++r) {
            float v = sacc[r];
            v += __shfl_xor(v, 1, 32);
            v += __shfl_xor(v, 2, 32);
            v += __shfl_xor(v, 4, 32);
            v += __shfl_xor(v, 8, 32);
            sacc[r] = v;
        }
        if ((lane & 15) == 0) {
            const int half = lane >> 4;   // C layout: lane<16 rows 0..7, lane>=16 rows 8..15
#pragma unroll
            for (int r = 0; r < 8; ++r) {
                int n = row0 + waveRow + half * 8 + r;
                float s = sacc[r] + brb[h];
                float m = mask[(size_t)b * N_ + n] ? s : -1.0e9f;
                size_t idx = ((size_t)b * H_ + h) * N_ + n;
                ws_scores[idx]  = m;
                out_scores[idx] = m;   // head_attentions = masked pre-softmax scores
            }
        }
    }
}

// ---------------------------------------------------------------------------
// K2: softmax over N per (b,h). 256 threads, 32 values/thread in registers.
// ---------------------------------------------------------------------------
__global__ void __launch_bounds__(256)
k_softmax(const float* __restrict__ scores, float* __restrict__ attn) {
    __shared__ float red[256];
    const int bh = blockIdx.x, t = threadIdx.x;
    const float* s = scores + (size_t)bh * N_;
    float vals[N_ / 256];
    float vmax = -3.4e38f;
#pragma unroll
    for (int i = 0; i < N_ / 256; ++i) { vals[i] = s[t + i * 256]; vmax = fmaxf(vmax, vals[i]); }
    red[t] = vmax; __syncthreads();
    for (int off = 128; off > 0; off >>= 1) {
        if (t < off) red[t] = fmaxf(red[t], red[t + off]);
        __syncthreads();
    }
    const float m = red[0]; __syncthreads();
    float sum = 0.f;
#pragma unroll
    for (int i = 0; i < N_ / 256; ++i) { vals[i] = __expf(vals[i] - m); sum += vals[i]; }
    red[t] = sum; __syncthreads();
    for (int off = 128; off > 0; off >>= 1) {
        if (t < off) red[t] += red[t + off];
        __syncthreads();
    }
    const float inv = 1.0f / red[0];
    float* o = attn + (size_t)bh * N_;
#pragma unroll
    for (int i = 0; i < N_ / 256; ++i) o[t + i * 256] = vals[i] * inv;
}

// ---------------------------------------------------------------------------
// K3: pooled partials over N-chunks (memory-bound second pass over features).
// ---------------------------------------------------------------------------
__global__ void __launch_bounds__(256)
k_pool_partial(const float* __restrict__ features, const float* __restrict__ attn,
               float* __restrict__ part) {
    __shared__ float satt[H_][N_ / NCH];
    const int bid  = blockIdx.x;
    const int dblk = bid % (D_ / 256);
    const int rest = bid / (D_ / 256);
    const int nc   = rest & (NCH - 1);
    const int b    = rest >> 3;
    const int t    = threadIdx.x;
    const int nlen = N_ / NCH;    // 1024

    for (int i = t; i < H_ * nlen; i += 256) {
        int h = i >> 10, n = i & (nlen - 1);
        satt[h][n] = attn[((size_t)b * H_ + h) * N_ + nc * nlen + n];
    }
    __syncthreads();

    const int d = dblk * 256 + t;
    float a0 = 0.f, a1 = 0.f, a2 = 0.f, a3 = 0.f;
    const float* f = features + ((size_t)b * N_ + (size_t)nc * nlen) * D_ + d;
    for (int n = 0; n < nlen; ++n) {
        float fv = f[(size_t)n * D_];
        a0 += satt[0][n] * fv; a1 += satt[1][n] * fv;
        a2 += satt[2][n] * fv; a3 += satt[3][n] * fv;
    }
    size_t base = (((size_t)b * NCH + nc) * H_) * D_ + d;
    part[base + 0 * D_] = a0; part[base + 1 * D_] = a1;
    part[base + 2 * D_] = a2; part[base + 3 * D_] = a3;
}

__global__ void __launch_bounds__(256)
k_pool_reduce(const float* __restrict__ part, float* __restrict__ pooled) {
    int idx = blockIdx.x * 256 + threadIdx.x;   // over B*H*D
    int d  = idx % D_;
    int bh = idx / D_;
    int b = bh / H_, h = bh % H_;
    float s = 0.f;
    for (int nc = 0; nc < NCH; ++nc)
        s += part[(((size_t)b * NCH + nc) * H_ + h) * D_ + d];
    pooled[idx] = s;
}

// ---------------------------------------------------------------------------
// K4: M = pooled.reshape(B, H*D) @ condW + condb  (pooled row cached in LDS)
// ---------------------------------------------------------------------------
__global__ void __launch_bounds__(256)
k_cond(const float* __restrict__ pooled, const float* __restrict__ condW,
       const float* __restrict__ condb, float* __restrict__ M) {
    __shared__ float sp[H_ * D_];
    const int b  = blockIdx.x / (D_ / 256);
    const int dc = (blockIdx.x % (D_ / 256)) * 256 + threadIdx.x;
    for (int i = threadIdx.x; i < H_ * D_; i += 256)
        sp[i] = pooled[(size_t)b * H_ * D_ + i];
    __syncthreads();
    float acc = condb[dc];
    for (int k = 0; k < H_ * D_; ++k)
        acc += sp[k] * condW[(size_t)k * D_ + dc];
    M[(size_t)b * D_ + dc] = acc;
}

// ---------------------------------------------------------------------------
// K5: per-sample head: sf = gelu(LN(M@sfW+sfb)); hp = gelu(te@h1W+h1b)@h2W+h2b
//     y = sum(sf*hp[:256]) + hp[256]
// ---------------------------------------------------------------------------
__global__ void __launch_bounds__(256)
k_head(const float* __restrict__ Mrow,
       const float* __restrict__ sfW, const float* __restrict__ sfb,
       const float* __restrict__ sfg, const float* __restrict__ sfbeta,
       const int* __restrict__ tissue, const float* __restrict__ temb,
       const float* __restrict__ h1W, const float* __restrict__ h1b,
       const float* __restrict__ h2W, const float* __restrict__ h2b,
       float* __restrict__ y) {
    __shared__ float sM[D_];
    __shared__ float red[256];
    __shared__ float act1[E_];
    __shared__ float s_hp256;
    const int b = blockIdx.x, t = threadIdx.x;
    for (int i = t; i < D_; i += 256) sM[i] = Mrow[(size_t)b * D_ + i];
    __syncthreads();

    float acc = sfb[t];
    for (int d = 0; d < D_; ++d) acc += sM[d] * sfW[(size_t)d * 256 + t];

    red[t] = acc; __syncthreads();
    for (int off = 128; off > 0; off >>= 1) { if (t < off) red[t] += red[t + off]; __syncthreads(); }
    const float mu = red[0] * (1.0f / 256.0f); __syncthreads();
    const float c = acc - mu;
    red[t] = c * c; __syncthreads();
    for (int off = 128; off > 0; off >>= 1) { if (t < off) red[t] += red[t + off]; __syncthreads(); }
    const float var = red[0] * (1.0f / 256.0f); __syncthreads();
    const float sh = gelu_f(c * rsqrtf(var + 1e-5f) * sfg[t] + sfbeta[t]);

    if (t < E_) {
        const int td = tissue[b];
        float a1 = h1b[t];
        for (int i = 0; i < E_; ++i) a1 += temb[(size_t)td * E_ + i] * h1W[(size_t)i * 64 + t];
        act1[t] = gelu_f(a1);
    }
    __syncthreads();

    float hp = h2b[t];
    for (int e = 0; e < E_; ++e) hp += act1[e] * h2W[(size_t)e * 257 + t];
    if (t == 0) {
        float v = h2b[256];
        for (int e = 0; e < E_; ++e) v += act1[e] * h2W[(size_t)e * 257 + 256];
        s_hp256 = v;
    }
    red[t] = sh * hp; __syncthreads();
    for (int off = 128; off > 0; off >>= 1) { if (t < off) red[t] += red[t + off]; __syncthreads(); }
    if (t == 0) y[b] = red[0] + s_hp256;
}

// ---------------------------------------------------------------------------
extern "C" void kernel_launch(void* const* d_in, const int* in_sizes, int n_in,
                              void* d_out, int out_size, void* d_ws, size_t ws_size,
                              hipStream_t stream) {
    const float*         features = (const float*)d_in[0];
    const unsigned char* mask     = (const unsigned char*)d_in[1];
    const int*           tissue   = (const int*)d_in[2];
    const float* attW   = (const float*)d_in[3];
    const float* attb   = (const float*)d_in[4];
    const float* gateW  = (const float*)d_in[5];
    const float* gateb  = (const float*)d_in[6];
    const float* brW    = (const float*)d_in[7];
    const float* brb    = (const float*)d_in[8];
    const float* condW  = (const float*)d_in[9];
    const float* condb  = (const float*)d_in[10];
    const float* sfW    = (const float*)d_in[11];
    const float* sfb    = (const float*)d_in[12];
    const float* sfg    = (const float*)d_in[13];
    const float* sfbeta = (const float*)d_in[14];
    const float* temb   = (const float*)d_in[15];
    const float* h1W    = (const float*)d_in[16];
    const float* h1b    = (const float*)d_in[17];
    const float* h2W    = (const float*)d_in[18];
    const float* h2b    = (const float*)d_in[19];

    // Workspace carve-up (bytes): Wb 6291456 | scores 1048576 | attn 1048576 |
    // part 1572864 | pooled 196608 | M 49152  (total ~9.7 MB)
    char*  ws        = (char*)d_ws;
    bf16*  Wb        = (bf16*)(ws);
    float* ws_scores = (float*)(ws + 6291456);
    float* ws_attn   = (float*)(ws + 7340032);
    float* ws_part   = (float*)(ws + 8388608);
    float* ws_pooled = (float*)(ws + 9961472);
    float* ws_M      = (float*)(ws + 10158080);

    float* y_out      = (float*)d_out;      // [8]
    float* scores_out = y_out + 8;          // [B,1,H,N] = [8,4,8192]

    k_prep_weights<<<(NCOL * D_) / 256, 256, 0, stream>>>(attW, gateW, Wb);

    // LDS is static inside k_scores (~259 KB): no dynamic shared mem needed.
    k_scores<<<B_ * (N_ / TILE_M), 128, 0, stream>>>(
        features, mask, Wb, attb, gateb, brW, brb, ws_scores, scores_out);

    k_softmax<<<B_ * H_, 256, 0, stream>>>(ws_scores, ws_attn);

    k_pool_partial<<<B_ * NCH * (D_ / 256), 256, 0, stream>>>(features, ws_attn, ws_part);
    k_pool_reduce<<<(B_ * H_ * D_) / 256, 256, 0, stream>>>(ws_part, ws_pooled);

    k_cond<<<B_ * (D_ / 256), 256, 0, stream>>>(ws_pooled, condW, condb, ws_M);

    k_head<<<B_, 256, 0, stream>>>(ws_M, sfW, sfb, sfg, sfbeta,
                                   tissue, temb, h1W, h1b, h2W, h2b, y_out);
}
